// EfficientDet_16716012716746
// MI455X (gfx1250) — compile-verified, matrix operand order checked by reading, not run on hardware
//
#include <hip/hip_runtime.h>
#include <hip/hip_bf16.h>
#include <stdint.h>

// ---------------------------------------------------------------------------
// EfficientDet post-process for MI455X (gfx1250, wave32).
// Pipeline:
//   0) zero histograms
//   1) per-level TDM-streamed 4096-bin histogram of logit keys      (pass 1)
//   2) per-image threshold-bin scan
//   3) per-level TDM-streamed compaction of candidates (bin >= t)   (pass 2)
//   4) exact 32-bit radix refinement -> exact top-5000 per image
//   5) WMMA-assisted box decode + sigmoid
//   6) per-image sequential Gaussian soft-NMS (100 rows)
// Workspace requirement: ~4.3 MB.
// ---------------------------------------------------------------------------

typedef __attribute__((ext_vector_type(16))) _Float16 v16h;
typedef __attribute__((ext_vector_type(8)))  float    v8f;
typedef __attribute__((ext_vector_type(4)))  unsigned u32x4;
typedef __attribute__((ext_vector_type(8)))  unsigned u32x8;

#define NIMG    16
#define NCLS    90
#define KTOP    5000
#define CAP     12288      // candidate capacity per image (>= KTOP + slack)
#define NBINS   4096       // 12-bit key histogram
#define UNIT_DW 6480       // dwords of cls data per block (25.9 KB LDS tile)
#define WAVE_DW 810        // dwords per wave-TDM descriptor (8 waves/block)
#define NDET    100
#define NEGV    (-1.0f)

// Monotone float -> uint key (larger float => larger key)
__device__ __forceinline__ unsigned fkey(float v) {
  unsigned u = __float_as_uint(v);
  return (u & 0x80000000u) ? ~u : (u | 0x80000000u);
}

// ---------------------------------------------------------------------------
// Tensor Data Mover: 1-D global->LDS DMA of `ndw` dwords.
// D# built per CDNA5 ISA (2 SGPR groups, 1-D tile).
// ---------------------------------------------------------------------------
__device__ __forceinline__ void tdm_load_1d(const float* gsrc, const float* ldst,
                                            unsigned ndw) {
  unsigned long long ga = (unsigned long long)(uintptr_t)gsrc;
  unsigned lds = (unsigned)(uintptr_t)ldst;   // low 32 bits of generic ptr = LDS offset
  u32x4 g0;
  g0[0] = 1u;                                  // count=1, is_restore=0, gather off
  g0[1] = lds;                                 // lds_addr
  g0[2] = (unsigned)ga;                        // global_addr[31:0]
  g0[3] = (unsigned)((ga >> 32) & 0x01FFFFFFull) | 0x80000000u; // ga[56:32] | type=2
  u32x8 g1;
  g1[0] = 0x00020000u;                         // data_size=2 (4 bytes), mask=0
  g1[1] = (ndw & 0xFFFFu) << 16;               // tensor_dim0[15:0]
  g1[2] = (ndw >> 16) | (1u << 16);            // tensor_dim0[31:16], tensor_dim1=1
  g1[3] = (ndw & 0xFFFFu) << 16;               // tile_dim0 = ndw
  g1[4] = 1u;                                  // tile_dim1=1, tile_dim2=0
  g1[5] = ndw;                                 // tensor_dim0_stride[31:0]
  g1[6] = 0u;
  g1[7] = 0u;
  asm volatile("tensor_load_to_lds %0, %1" :: "s"(g0), "s"(g1) : "memory");
}

__global__ void zero_u32_kernel(unsigned* __restrict__ p, int n) {
  int i = blockIdx.x * blockDim.x + threadIdx.x;
  if (i < n) p[i] = 0u;
}

// ---------------------------------------------------------------------------
// Streaming pass over one cls level. MODE 0: histogram. MODE 1: compaction.
// Block = 256 threads (8 waves); each wave TDM-loads a 810-dword slice.
// ---------------------------------------------------------------------------
template <int MODE>
__global__ void cls_scan_kernel(const float* __restrict__ cls,
                                int slice_dw, int hw_log2, int anchor_off,
                                unsigned* __restrict__ hist,
                                const unsigned* __restrict__ thr,
                                unsigned* __restrict__ cntA,
                                float* __restrict__ cand_val,
                                unsigned* __restrict__ cand_idx) {
  __shared__ float    tile[UNIT_DW];
  __shared__ unsigned lhist[NBINS];
  const int b    = blockIdx.y;
  const int tid  = threadIdx.x;
  const int lane = tid & 31;
  const int wid  = __builtin_amdgcn_readfirstlane(tid >> 5);

  if (MODE == 0) {
    for (int i = tid; i < NBINS; i += 256) lhist[i] = 0u;
    __syncthreads();
  }

  const size_t base = (size_t)b * (size_t)slice_dw +
                      (size_t)blockIdx.x * UNIT_DW + (size_t)wid * WAVE_DW;
  tdm_load_1d(cls + base, &tile[wid * WAVE_DW], WAVE_DW);

  unsigned tbin = 0;
  if (MODE == 1) tbin = thr[b * 4];

  __builtin_amdgcn_s_wait_tensorcnt(0);
  asm volatile("" ::: "memory");

  for (int j = lane; j < WAVE_DW; j += 32) {
    float v = tile[wid * WAVE_DW + j];
    unsigned bin = fkey(v) >> 20;
    if (MODE == 0) {
      atomicAdd(&lhist[bin], 1u);
    } else if (bin >= tbin) {
      unsigned p = atomicAdd(&cntA[b], 1u);
      if (p < CAP) {
        int idx  = blockIdx.x * UNIT_DW + wid * WAVE_DW + j; // flat [810,H,W] index
        int ch   = idx >> hw_log2;
        int rest = idx & ((1 << hw_log2) - 1);               // y*W + x
        int a    = ch / NCLS;
        int c    = ch - a * NCLS;
        unsigned n = (unsigned)(anchor_off + rest * 9 + a);  // global anchor id (<49104)
        cand_val[(size_t)b * CAP + p] = v;
        cand_idx[(size_t)b * CAP + p] = ((unsigned)c << 16) | n;
      }
    }
  }

  if (MODE == 0) {
    __syncthreads();
    for (int i = tid; i < NBINS; i += 256) {
      unsigned h = lhist[i];
      if (h) atomicAdd(&hist[b * NBINS + i], h);
    }
  }
}

// Per-image: find the histogram bin where the suffix count crosses KTOP.
__global__ void select_threshold_kernel(const unsigned* __restrict__ hist,
                                        unsigned* __restrict__ thr,
                                        unsigned* __restrict__ cntA) {
  __shared__ unsigned h[NBINS];
  const int b = blockIdx.x;
  for (int i = threadIdx.x; i < NBINS; i += 256) h[i] = hist[b * NBINS + i];
  __syncthreads();
  if (threadIdx.x == 0) {
    unsigned cum = 0; int t = 0;
    for (int bin = NBINS - 1; bin >= 0; --bin) {
      unsigned c = h[bin];
      if (cum + c >= KTOP) { t = bin; break; }
      cum += c;
    }
    thr[b * 4] = (unsigned)t;
    cntA[b]    = 0u;
  }
}

// ---------------------------------------------------------------------------
// Exact top-KTOP among candidates: 4-round byte radix refinement of the
// 32-bit key, then deterministic boundary handling (rank equal-key elements
// by packed (class,anchor) index, matching stable lax.top_k tie order).
// ---------------------------------------------------------------------------
__global__ void exact_select_kernel(const float* __restrict__ cand_val,
                                    const unsigned* __restrict__ cand_idx,
                                    const unsigned* __restrict__ cntA,
                                    float* __restrict__ out_val,
                                    unsigned* __restrict__ out_idx) {
  __shared__ unsigned h[256];
  __shared__ unsigned s_prefix, s_krem, s_ctr;
  const int b = blockIdx.x, tid = threadIdx.x;
  int nc = (int)cntA[b]; if (nc > CAP) nc = CAP;
  const float*    cv = cand_val + (size_t)b * CAP;
  const unsigned* ci = cand_idx + (size_t)b * CAP;

  // Sentinel init (score ~0) so all KTOP slots are defined every call.
  for (int j = tid; j < KTOP; j += 256) {
    out_val[(size_t)b * KTOP + j] = -40.0f;
    out_idx[(size_t)b * KTOP + j] = 0u;
  }
  if (tid == 0) { s_prefix = 0u; s_krem = KTOP; }
  __syncthreads();

  for (int round = 3; round >= 0; --round) {
    h[tid & 255] = 0u;
    __syncthreads();
    const int shift = round * 8;
    const unsigned hiMask = (round == 3) ? 0u : (0xFFFFFFFFu << (shift + 8));
    const unsigned pref = s_prefix;
    const unsigned krem = s_krem;
    for (int j = tid; j < nc; j += 256) {
      unsigned key = fkey(cv[j]);
      if ((key & hiMask) == (pref & hiMask))
        atomicAdd(&h[(key >> shift) & 0xFFu], 1u);
    }
    __syncthreads();
    if (tid == 0) {
      unsigned cum = 0; int vb = 0;
      for (int v = 255; v >= 1; --v) {
        unsigned c = h[v];
        if (cum + c >= krem) { vb = v; break; }
        cum += c;
      }
      s_prefix = pref | ((unsigned)vb << shift);
      s_krem   = krem - cum;
    }
    __syncthreads();
  }

  const unsigned keyT   = s_prefix;
  const unsigned needEq = s_krem;
  if (tid == 0) s_ctr = 0u;
  __syncthreads();

  for (int j = tid; j < nc; j += 256) {
    unsigned key = fkey(cv[j]);
    if (key > keyT) {
      unsigned slot = atomicAdd(&s_ctr, 1u);
      if (slot < KTOP - needEq) {
        out_val[(size_t)b * KTOP + slot] = cv[j];
        out_idx[(size_t)b * KTOP + slot] = ci[j];
      }
    } else if (key == keyT) {
      unsigned mine = ci[j], rank = 0;
      for (int jj = 0; jj < nc; ++jj)
        if (fkey(cv[jj]) == keyT && ci[jj] < mine) rank++;
      if (rank < needEq) {
        unsigned slot = KTOP - needEq + rank;
        out_val[(size_t)b * KTOP + slot] = cv[j];
        out_idx[(size_t)b * KTOP + slot] = ci[j];
      }
    }
  }
}

// ---------------------------------------------------------------------------
// Box decode. Each wave handles 16 boxes; the 4x4 anchor transform
// [ya,xa,ha,wa] = M * [y1,x1,y2,x2] is computed as one v_wmma_f32_16x16x32_f16
// (A = anchor rows in K=0..3, B = constant 4x16 matrix, K=4..31 zero).
// ---------------------------------------------------------------------------
__global__ void decode_kernel(const float* __restrict__ cand_val,
                              const unsigned* __restrict__ cand_idx,
                              const float* __restrict__ anchors,
                              const float* __restrict__ b0, const float* __restrict__ b1,
                              const float* __restrict__ b2, const float* __restrict__ b3,
                              const float* __restrict__ b4,
                              float* __restrict__ dec_boxes,
                              float* __restrict__ dec_scores,
                              float* __restrict__ dec_cls) {
  __shared__ float q[8][16][4];
  const int b    = blockIdx.y;
  const int tid  = threadIdx.x;
  const int lane = tid & 31;
  const int wid  = tid >> 5;
  const int g    = (blockIdx.x * 8 + wid) * 16;   // wave's 16-box group base
  if (g >= KTOP) return;                          // uniform per wave

  // --- A matrix: lane<16 supplies row m=lane, K=0..3 = anchor yxyx ---
  int m = lane & 15;
  int i = g + m; if (i >= KTOP) i = KTOP - 1;
  unsigned pack = cand_idx[(size_t)b * KTOP + i];
  unsigned n    = pack & 0xFFFFu;
  v16h A = {};
  v16h Bm = {};
  if (lane < 16) {
    const float4 ar = *(const float4*)(anchors + 4 * (size_t)n); // y1 x1 y2 x2
    A[0] = (_Float16)ar.x; A[1] = (_Float16)ar.y;
    A[2] = (_Float16)ar.z; A[3] = (_Float16)ar.w;
    // B[k][n] (k=0..3), columns: 0->ya 1->xa 2->ha 3->wa
    if (lane == 0)      { Bm[0] = (_Float16)0.5f;  Bm[2] = (_Float16)0.5f; }
    else if (lane == 1) { Bm[1] = (_Float16)0.5f;  Bm[3] = (_Float16)0.5f; }
    else if (lane == 2) { Bm[0] = (_Float16)-1.0f; Bm[2] = (_Float16)1.0f; }
    else if (lane == 3) { Bm[1] = (_Float16)-1.0f; Bm[3] = (_Float16)1.0f; }
  }
  v8f C = {};
  v8f D = __builtin_amdgcn_wmma_f32_16x16x32_f16(false, A, false, Bm,
                                                 (short)0, C, false, false);
  // D[m][ncol]: lanes 0-15 hold rows r (boxes 0-7), lanes 16-31 rows 8+r.
  int ncol = lane & 15;
  if (ncol < 4) {
    int mbase = (lane < 16) ? 0 : 8;
#pragma unroll
    for (int r = 0; r < 8; ++r) q[wid][mbase + r][ncol] = D[r];
  }
  asm volatile("s_wait_dscnt 0" ::: "memory");   // intra-wave LDS RAW

  if (lane < 16 && (g + lane) < KTOP) {
    const int i2 = g + lane;
    unsigned pk = cand_idx[(size_t)b * KTOP + i2];
    unsigned nn = pk & 0xFFFFu, cc = pk >> 16;
    float ya = q[wid][lane][0], xa = q[wid][lane][1];
    float ha = q[wid][lane][2], wa = q[wid][lane][3];

    int off, Wl; const float* bp;
    if (nn < 36864u)      { off = 0;     Wl = 64; bp = b0; }
    else if (nn < 46080u) { off = 36864; Wl = 32; bp = b1; }
    else if (nn < 48384u) { off = 46080; Wl = 16; bp = b2; }
    else if (nn < 48960u) { off = 48384; Wl = 8;  bp = b3; }
    else                  { off = 48960; Wl = 4;  bp = b4; }
    int local = (int)nn - off;
    int cell  = local / 9;
    int a     = local - cell * 9;
    int HWl   = Wl * Wl;
    size_t base = ((size_t)(b * 36 + a * 4)) * (size_t)HWl + (size_t)cell;
    float ty = bp[base], tx = bp[base + HWl];
    float th = bp[base + 2 * (size_t)HWl], tw = bp[base + 3 * (size_t)HWl];

    float hb = expf(th) * ha, wb = expf(tw) * wa;
    float yc = ty * ha + ya,  xc = tx * wa + xa;
    float4 ob = make_float4(xc - 0.5f * wb, yc - 0.5f * hb,
                            xc + 0.5f * wb, yc + 0.5f * hb);
    *(float4*)(dec_boxes + ((size_t)b * KTOP + i2) * 4) = ob;
    float v = cand_val[(size_t)b * KTOP + i2];
    dec_scores[(size_t)b * KTOP + i2] = 1.0f / (1.0f + expf(-v));
    dec_cls[(size_t)b * KTOP + i2]    = (float)cc;
  }
}

// ---------------------------------------------------------------------------
// Gaussian soft-NMS, one image per block (512 threads = 16 waves).
// Scores in LDS; boxes stay L2-resident in global.
// ---------------------------------------------------------------------------
__global__ void softnms_kernel(const float* __restrict__ dec_boxes,
                               const float* __restrict__ dec_scores,
                               const float* __restrict__ dec_cls,
                               float* __restrict__ out) {
  __shared__ float sc[KTOP];
  __shared__ float red_v[16];
  __shared__ int   red_i[16];
  __shared__ float s_selv, s_box[4];
  __shared__ int   s_seli;
  const int b   = blockIdx.x;
  const int tid = threadIdx.x;
  const float* bxs = dec_boxes + (size_t)b * KTOP * 4;

  for (int i = tid; i < KTOP; i += 512) sc[i] = dec_scores[(size_t)b * KTOP + i];
  __syncthreads();

  for (int t = 0; t < NDET; ++t) {
    // --- block argmax (first index on ties, matching jnp.argmax) ---
    float v = -1e30f; int idx = 0;
    for (int i = tid; i < KTOP; i += 512) {
      float s = sc[i];
      if (s > v) { v = s; idx = i; }
    }
    for (int off = 16; off > 0; off >>= 1) {
      float ov = __shfl_down(v, off);
      int   oi = __shfl_down(idx, off);
      if (ov > v || (ov == v && oi < idx)) { v = ov; idx = oi; }
    }
    if ((tid & 31) == 0) { red_v[tid >> 5] = v; red_i[tid >> 5] = idx; }
    __syncthreads();
    if (tid == 0) {
      float bv = red_v[0]; int bi = red_i[0];
      for (int w = 1; w < 16; ++w) {
        float ov = red_v[w]; int oi = red_i[w];
        if (ov > bv || (ov == bv && oi < bi)) { bv = ov; bi = oi; }
      }
      s_selv = bv; s_seli = bi;
      const float* bb = bxs + (size_t)bi * 4;
      s_box[0] = bb[0]; s_box[1] = bb[1]; s_box[2] = bb[2]; s_box[3] = bb[3];
    }
    __syncthreads();

    const float bx1 = s_box[0], by1 = s_box[1], bx2 = s_box[2], by2 = s_box[3];
    const int   sel = s_seli;
    const float sval = s_selv;
    const float areaA = fmaxf(bx2 - bx1, 0.f) * fmaxf(by2 - by1, 0.f);

    for (int i = tid; i < KTOP; i += 512) {
      float4 bo = *(const float4*)(bxs + (size_t)i * 4);
      float ix1 = fmaxf(bx1, bo.x), iy1 = fmaxf(by1, bo.y);
      float ix2 = fminf(bx2, bo.z), iy2 = fminf(by2, bo.w);
      float inter = fmaxf(ix2 - ix1, 0.f) * fmaxf(iy2 - iy1, 0.f);
      float areaB = fmaxf(bo.z - bo.x, 0.f) * fmaxf(bo.w - bo.y, 0.f);
      float iou = inter / fmaxf(areaA + areaB - inter, 1e-8f);
      float s = sc[i] * expf(-2.0f * iou * iou);   // sigma = 0.5
      if (i == sel || !(s > 0.005f)) s = NEGV;
      sc[i] = s;
    }
    if (tid == 0) {
      float* row = out + ((size_t)b * NDET + t) * 6;
      bool valid = sval > 0.005f;
      row[0] = valid ? bx1 : 0.f; row[1] = valid ? by1 : 0.f;
      row[2] = valid ? bx2 : 0.f; row[3] = valid ? by2 : 0.f;
      row[4] = valid ? sval : 0.f;
      row[5] = valid ? (dec_cls[(size_t)b * KTOP + sel] + 1.0f) : 0.f;
    }
    __syncthreads();
  }
}

// ---------------------------------------------------------------------------
extern "C" void kernel_launch(void* const* d_in, const int* in_sizes, int n_in,
                              void* d_out, int out_size, void* d_ws, size_t ws_size,
                              hipStream_t stream) {
  const float* cls[5]; const float* box[5];
  for (int l = 0; l < 5; ++l) {
    cls[l] = (const float*)d_in[2 * l];
    box[l] = (const float*)d_in[2 * l + 1];
  }
  const float* anchors = (const float*)d_in[10];

  // workspace layout (dword offsets); total ~4.3 MB
  unsigned* ws32 = (unsigned*)d_ws;
  unsigned* hist = ws32;                        // 16*4096
  unsigned* thr  = ws32 + 65536;                // 16*4
  unsigned* cntA = ws32 + 65600;                // 16
  float*    cval = (float*)(ws32 + 65632);      // 16*CAP
  unsigned* cidx = ws32 + 262240;               // 16*CAP
  float*    c2v  = (float*)(ws32 + 458848);     // 16*KTOP
  unsigned* c2i  = ws32 + 538848;               // 16*KTOP
  float*    dbox = (float*)(ws32 + 618848);     // 16*KTOP*4
  float*    dsc  = (float*)(ws32 + 938848);     // 16*KTOP
  float*    dcl  = (float*)(ws32 + 1018848);    // 16*KTOP

  static const int kW[5]    = {64, 32, 16, 8, 4};
  static const int kHWlog[5]= {12, 10, 8, 6, 4};
  static const int kAoff[5] = {0, 36864, 46080, 48384, 48960};

  zero_u32_kernel<<<256, 256, 0, stream>>>(hist, NIMG * NBINS);

  for (int l = 0; l < 5; ++l) {
    int slice = 810 * kW[l] * kW[l];
    dim3 grid(slice / UNIT_DW, NIMG);
    cls_scan_kernel<0><<<grid, 256, 0, stream>>>(cls[l], slice, kHWlog[l], kAoff[l],
                                                 hist, nullptr, nullptr, nullptr, nullptr);
  }

  select_threshold_kernel<<<NIMG, 256, 0, stream>>>(hist, thr, cntA);

  for (int l = 0; l < 5; ++l) {
    int slice = 810 * kW[l] * kW[l];
    dim3 grid(slice / UNIT_DW, NIMG);
    cls_scan_kernel<1><<<grid, 256, 0, stream>>>(cls[l], slice, kHWlog[l], kAoff[l],
                                                 nullptr, thr, cntA, cval, cidx);
  }

  exact_select_kernel<<<NIMG, 256, 0, stream>>>(cval, cidx, cntA, c2v, c2i);

  dim3 dgrid((KTOP + 127) / 128, NIMG);   // 8 waves * 16 boxes per block
  decode_kernel<<<dgrid, 256, 0, stream>>>(c2v, c2i, anchors,
                                           box[0], box[1], box[2], box[3], box[4],
                                           dbox, dsc, dcl);

  softnms_kernel<<<NIMG, 512, 0, stream>>>(dbox, dsc, dcl, (float*)d_out);
}
